// KMEANS_67980742361662
// MI455X (gfx1250) — compile-verified, matrix operand order checked by reading, not run on hardware
//
#include <hip/hip_runtime.h>
#include <hip/hip_bf16.h>

// ---------------------------------------------------------------------------
// KMEANS distance/argmin/loss/acc for MI455X (gfx1250, wave32, WMMA).
// GEMM x@centers^T done with v_wmma_f32_16x16x32_bf16 using a bf16 hi/lo
// split (3 WMMAs ~ fp32 accuracy for argmin purposes); x2 / c2 terms exact fp32.
// ---------------------------------------------------------------------------

typedef __attribute__((ext_vector_type(16))) __bf16 v16bf;
typedef __attribute__((ext_vector_type(8)))  float  v8f;

#define DIMD 128
#define KC   256
#define NCLS 10
#define ROWS_PER_WG 128
#define WAVES 8

union FragBF {
    v16bf v;
    unsigned short s[16];
    uint4 u[2];
};

__device__ __forceinline__ unsigned short f2bf(float f) {
    unsigned u = __float_as_uint(f);
    u = u + 0x7FFFu + ((u >> 16) & 1u);   // round-to-nearest-even
    return (unsigned short)(u >> 16);
}
__device__ __forceinline__ float bf2f(unsigned short h) {
    return __uint_as_float(((unsigned)h) << 16);
}

// --- zero the global confusion matrix (d_ws is not re-poisoned between runs)
__global__ void kmeans_zero(int* __restrict__ conf_g) {
    int i = blockIdx.x * blockDim.x + threadIdx.x;
    if (i < KC * NCLS) conf_g[i] = 0;
}

// --- convert centers fp32 -> bf16 hi/lo, compute exact |c|^2 per center
__global__ void kmeans_prep(const float* __restrict__ centers,
                            unsigned short* __restrict__ chi,
                            unsigned short* __restrict__ clo,
                            float* __restrict__ csq) {
    int c = threadIdx.x;                  // 1 block x 256 threads, 1 center/thread
    const float* row = centers + (size_t)c * DIMD;
    float sq = 0.f;
    for (int d = 0; d < DIMD; ++d) {
        float v = row[d];
        sq += v * v;
        unsigned short h = f2bf(v);
        chi[c * DIMD + d] = h;
        clo[c * DIMD + d] = f2bf(v - bf2f(h));
    }
    csq[c] = sq;
}

// --- main: per 16-row tile per wave, WMMA over all K=256 centers
__global__ __launch_bounds__(256)
void kmeans_main(const float* __restrict__ x,
                 const int* __restrict__ y,
                 const unsigned short* __restrict__ chi,
                 const unsigned short* __restrict__ clo,
                 const float* __restrict__ csq,
                 float* __restrict__ partial_loss,
                 int* __restrict__ conf_g) {
    __shared__ int   conf_s[KC * NCLS];
    __shared__ float loss_s[WAVES];

    const int tid = threadIdx.x;
    for (int i = tid; i < KC * NCLS; i += 256) conf_s[i] = 0;
    if (tid < WAVES) loss_s[tid] = 0.f;
    __syncthreads();

    const int lane = tid & 31;
    const int wave = tid >> 5;
    const int m    = lane & 15;       // A: row in tile / B: column in tile
    const int hi   = lane >> 4;       // half-wave select
    const int ko   = hi * 8;          // ISA 16-bit A/B fragment K packing
    const int rowBase = blockIdx.x * ROWS_PER_WG + wave * 16;

    const float* xrow = x + (size_t)(rowBase + m) * DIMD;

    v8f acc[16];
#pragma unroll
    for (int t = 0; t < 16; ++t) acc[t] = v8f{0,0,0,0,0,0,0,0};

    float sq = 0.f;   // this lane's share of |x_row|^2

    for (int ks = 0; ks < 4; ++ks) {
        const int base = ks * 32 + ko;
        // A fragment: K = base..base+7 and base+16..base+23 of this row
        float4 f0 = *(const float4*)(xrow + base);
        float4 f1 = *(const float4*)(xrow + base + 4);
        float4 f2 = *(const float4*)(xrow + base + 16);
        float4 f3 = *(const float4*)(xrow + base + 20);
        float fv[16] = {f0.x,f0.y,f0.z,f0.w, f1.x,f1.y,f1.z,f1.w,
                        f2.x,f2.y,f2.z,f2.w, f3.x,f3.y,f3.z,f3.w};
        FragBF ah, al;
#pragma unroll
        for (int e = 0; e < 16; ++e) {
            float v = fv[e];
            sq += v * v;
            unsigned short h = f2bf(v);
            ah.s[e] = h;
            al.s[e] = f2bf(v - bf2f(h));
        }
#pragma unroll
        for (int nt = 0; nt < 16; ++nt) {
            // B fragment: column n = nt*16 + m is centers row c, same K packing
            const uint4* bh = (const uint4*)(chi + ((size_t)(nt * 16 + m) * DIMD + base));
            const uint4* bl = (const uint4*)(clo + ((size_t)(nt * 16 + m) * DIMD + base));
            FragBF bhf, blf;
            bhf.u[0] = bh[0]; bhf.u[1] = bh[2];   // +16 elements = +2 uint4
            blf.u[0] = bl[0]; blf.u[1] = bl[2];
            acc[nt] = __builtin_amdgcn_wmma_f32_16x16x32_bf16(
                false, ah.v, false, bhf.v, (short)0, acc[nt], false, false);
            acc[nt] = __builtin_amdgcn_wmma_f32_16x16x32_bf16(
                false, ah.v, false, blf.v, (short)0, acc[nt], false, false);
            acc[nt] = __builtin_amdgcn_wmma_f32_16x16x32_bf16(
                false, al.v, false, bhf.v, (short)0, acc[nt], false, false);
        }
    }

    // full |x_row|^2: lanes L and L^16 cover complementary K halves
    float sqf = sq + __shfl_xor(sq, 16, 32);
    // C layout: lane holds col m, rows hi*8 + r; fetch matching row norms
    float xsq_r[8];
#pragma unroll
    for (int r = 0; r < 8; ++r) xsq_r[r] = __shfl(sqf, hi * 8 + r, 32);

    float best[8];
    int   bidx[8];
#pragma unroll
    for (int r = 0; r < 8; ++r) { best[r] = 3.4e38f; bidx[r] = 0; }

#pragma unroll
    for (int nt = 0; nt < 16; ++nt) {
        int col = nt * 16 + m;
        float cs = csq[col];
#pragma unroll
        for (int r = 0; r < 8; ++r) {
            float dd = xsq_r[r] - 2.f * acc[nt][r] + cs;
            if (dd < best[r]) { best[r] = dd; bidx[r] = col; }
        }
    }

    // min/argmin across the 16 lanes that share the same rows (xor <= 8 stays
    // inside each 16-lane column group of the C layout)
#pragma unroll
    for (int off = 8; off >= 1; off >>= 1) {
#pragma unroll
        for (int r = 0; r < 8; ++r) {
            float od = __shfl_xor(best[r], off, 32);
            int   oi = __shfl_xor(bidx[r], off, 32);
            if (od < best[r] || (od == best[r] && oi < bidx[r])) {
                best[r] = od; bidx[r] = oi;
            }
        }
    }

    float lpart = 0.f;
    if (m == 0) {                        // lanes 0 (rows 0-7) and 16 (rows 8-15)
#pragma unroll
        for (int r = 0; r < 8; ++r) {
            int grow = rowBase + hi * 8 + r;
            lpart += best[r];
            int lbl = y[grow];
            atomicAdd(&conf_s[bidx[r] * NCLS + lbl], 1);
        }
    }
    lpart += __shfl_xor(lpart, 16, 32);
    if (lane == 0) loss_s[wave] = lpart;
    __syncthreads();

    if (tid == 0) {
        float s = 0.f;
        for (int w = 0; w < WAVES; ++w) s += loss_s[w];
        partial_loss[blockIdx.x] = s;
    }
    for (int i = tid; i < KC * NCLS; i += 256) {
        int v = conf_s[i];
        if (v) atomicAdd(&conf_g[i], v);
    }
}

// --- finalize: ordered loss sum + per-cluster max-count -> accuracy
__global__ void kmeans_finalize(const float* __restrict__ partial_loss, int nPart,
                                const int* __restrict__ conf_g,
                                float* __restrict__ out, int N) {
    __shared__ float ls[256];
    __shared__ int   cs[256];
    int t = threadIdx.x;
    float lsum = 0.f;
    for (int i = t; i < nPart; i += 256) lsum += partial_loss[i];
    ls[t] = lsum;
    int mx = 0;
    for (int j = 0; j < NCLS; ++j) {
        int v = conf_g[t * NCLS + j];
        mx = v > mx ? v : mx;
    }
    cs[t] = mx;
    __syncthreads();
    for (int s2 = 128; s2 > 0; s2 >>= 1) {
        if (t < s2) { ls[t] += ls[t + s2]; cs[t] += cs[t + s2]; }
        __syncthreads();
    }
    if (t == 0) {
        out[0] = ls[0];
        out[1] = (float)cs[0] / (float)N;
    }
}

extern "C" void kernel_launch(void* const* d_in, const int* in_sizes, int n_in,
                              void* d_out, int out_size, void* d_ws, size_t ws_size,
                              hipStream_t stream) {
    const float* x       = (const float*)d_in[0];
    const int*   y       = (const int*)d_in[1];
    const float* centers = (const float*)d_in[2];
    const int N   = in_sizes[1];          // 262144
    const int nWG = N / ROWS_PER_WG;      // 2048

    // workspace layout (all 256B-aligned)
    char* ws = (char*)d_ws;
    unsigned short* chi     = (unsigned short*)(ws);                       // 64 KB
    unsigned short* clo     = (unsigned short*)(ws + 65536);               // 64 KB
    float*          csq     = (float*)(ws + 131072);                       // 1 KB
    float*          partial = (float*)(ws + 131072 + 4096);                // nWG*4
    int*            conf_g  = (int*)(ws + 131072 + 4096 + ((nWG * 4 + 255) & ~255));

    kmeans_zero<<<(KC * NCLS + 255) / 256, 256, 0, stream>>>(conf_g);
    kmeans_prep<<<1, 256, 0, stream>>>(centers, chi, clo, csq);
    kmeans_main<<<nWG, 256, 0, stream>>>(x, y, chi, clo, csq, partial, conf_g);
    kmeans_finalize<<<1, 256, 0, stream>>>(partial, nWG, conf_g, (float*)d_out, N);
}
